// Bi_IDMB_Patch_13941463843469
// MI455X (gfx1250) — matrix operand order repaired and checked by reading, not compiled, not run on hardware
//
#include <hip/hip_runtime.h>
#include <hip/hip_bf16.h>
#include <math.h>

typedef __attribute__((ext_vector_type(16))) _Float16 v16h;
typedef __attribute__((ext_vector_type(8)))  _Float16 v8h;
typedef __attribute__((ext_vector_type(8)))  float    v8f;

#define D_MODEL 256
#define D_STATE 32
#define D_INNER 512
#define DT_RANK 16
#define BATCH   8
#define SEQ     512
#define ROWS    (BATCH*SEQ)            // 4096
#define XDBL_W  (DT_RANK + 2*D_STATE)  // 80
#define CHUNKS  8
#define CLEN    (SEQ/CHUNKS)           // 64
#define GEMM_WAVES 4

// ---------------------------------------------------------------------------
// LayerNorm: one block per row of 256, write f32 and f16 copies
// ---------------------------------------------------------------------------
__global__ void ln_kernel(const float* __restrict__ x, const float* __restrict__ g,
                          const float* __restrict__ bb, float* __restrict__ xn,
                          _Float16* __restrict__ xnh) {
  __shared__ float red[D_MODEL];
  int row = blockIdx.x;
  int c = threadIdx.x;
  float v = x[(size_t)row*D_MODEL + c];
  red[c] = v; __syncthreads();
  for (int s = D_MODEL/2; s > 0; s >>= 1) { if (c < s) red[c] += red[c+s]; __syncthreads(); }
  float mu = red[0] * (1.0f/D_MODEL); __syncthreads();
  float d = v - mu;
  red[c] = d*d; __syncthreads();
  for (int s = D_MODEL/2; s > 0; s >>= 1) { if (c < s) red[c] += red[c+s]; __syncthreads(); }
  float var = red[0] * (1.0f/D_MODEL);
  float o = d * rsqrtf(var + 1e-5f) * g[c] + bb[c];
  xn [(size_t)row*D_MODEL + c] = o;
  xnh[(size_t)row*D_MODEL + c] = (_Float16)o;
}

__global__ void f2h_kernel(const float* __restrict__ src, _Float16* __restrict__ dst, int n) {
  int i = blockIdx.x*blockDim.x + threadIdx.x;
  if (i < n) dst[i] = (_Float16)src[i];
}

// ---------------------------------------------------------------------------
// Async-staged WMMA GEMM (NB tiles of N per block-strip, N % (16*NB) == 0,
// M % (16*GEMM_WAVES) == 0). All 4 waves of a block share one B strip:
// B fragments are staged global->LDS once per block per k-step with
// tensor-style async copies (global_load_async_to_lds_b128, ASYNCcnt),
// double-buffered against the WMMA work. A fragments stay per-wave global
// loads (unique rows). 48B/thread LDS slots keep 16B alignment and break
// the 32B-stride bank pattern.
// ---------------------------------------------------------------------------
template<int NB>
__global__ __launch_bounds__(32*GEMM_WAVES)
void wmma_gemm_async(const _Float16* __restrict__ A, const _Float16* __restrict__ W,
                     float* __restrict__ C, const float* __restrict__ bias,
                     int M, int N, int K) {
  constexpr int SLOT  = 48;                       // bytes per staged 32B slice
  constexpr int STAGE = 32*GEMM_WAVES*SLOT;       // bytes per buffer
  __shared__ __align__(32) char Bs[2*STAGE];

  int t    = threadIdx.x;
  int wid  = t >> 5;
  int lane = t & 31;
  int l15  = lane & 15;
  bool hi  = lane >= 16;
  int nblk = N/(16*NB);
  int nt   = blockIdx.x % nblk;
  int mt   = (blockIdx.x / nblk)*GEMM_WAVES + wid;

  // staging role: thread t = j*32 + tl stages the 32B slice that compute-lane
  // tl of fragment j will read (identical (l15,hi) mapping on both sides).
  int jj = t >> 5, tl = t & 31;
  const _Float16* bsrc = W + (size_t)((nt*NB + jj)*16 + (tl & 15))*K + ((tl >= 16) ? 16 : 0);
  const _Float16* Ap   = A + (size_t)((mt << 4) + l15)*K + (hi ? 16 : 0);
  __builtin_prefetch(Ap, 0, 0);                   // global_prefetch_b8

  // low 32 bits of a generic LDS pointer == byte offset from the wave's LDS base
  unsigned lds_base = (unsigned)(size_t)(void*)Bs;
  unsigned myslot   = lds_base + (unsigned)t * SLOT;

  auto stage = [&](int buf, int k0) {
    const _Float16* g = bsrc + k0;
    unsigned l = myslot + (unsigned)(buf*STAGE);
    // INST_OFFSET is added to BOTH lds and memory addresses (ISA 15.18.3)
    asm volatile("global_load_async_to_lds_b128 %0, %1, off"
                 :: "v"(l), "v"(g) : "memory");
    asm volatile("global_load_async_to_lds_b128 %0, %1, off offset:16"
                 :: "v"(l), "v"(g) : "memory");
  };

  stage(0, 0);
  v8f acc[NB] = {};
  int nsteps = K >> 5;
  for (int s = 0; s < nsteps; ++s) {
    int k0  = s << 5;
    int buf = s & 1;
    if (s + 1 < nsteps) {
      stage(buf ^ 1, k0 + 32);
      asm volatile("s_wait_asynccnt 0x2" ::: "memory");  // current buffer landed
    } else {
      asm volatile("s_wait_asynccnt 0x0" ::: "memory");
    }
    __syncthreads();                                     // cross-wave visibility
    v8h a0 = *(const v8h*)(Ap + k0);
    v8h a1 = *(const v8h*)(Ap + k0 + 8);
    v16h a;
#pragma unroll
    for (int i = 0; i < 8; ++i) { a[i] = a0[i]; a[8+i] = a1[i]; }
#pragma unroll
    for (int j = 0; j < NB; ++j) {
      const char* p = Bs + buf*STAGE + (size_t)(j*32 + lane)*SLOT;
      v8h b0 = *(const v8h*)p;
      v8h b1 = *(const v8h*)(p + 16);
      v16h b;
#pragma unroll
      for (int i = 0; i < 8; ++i) { b[i] = b0[i]; b[8+i] = b1[i]; }
      acc[j] = __builtin_amdgcn_wmma_f32_16x16x32_f16(
          false, a, false, b, (short)0, acc[j], false, false);
    }
    __syncthreads();                                     // before buffer reuse
  }
#pragma unroll
  for (int j = 0; j < NB; ++j) {
    int ncol = ((nt*NB + j) << 4) + l15;
    float bv = bias ? bias[ncol] : 0.0f;
#pragma unroll
    for (int r = 0; r < 8; ++r) {
      int row = (mt << 4) + r + (hi ? 8 : 0);
      C[(size_t)row * N + ncol] = acc[j][r] + bv;
    }
  }
}

// ---------------------------------------------------------------------------
// Direct-global register-blocked WMMA GEMM (used for N=80: NB=5, one wave
// covers the whole output row strip).
// ---------------------------------------------------------------------------
template<int NB>
__global__ void wmma_gemm(const _Float16* __restrict__ A, const _Float16* __restrict__ W,
                          float* __restrict__ C, const float* __restrict__ bias,
                          int M, int N, int K) {
  int wave = blockIdx.x * (blockDim.x >> 5) + (threadIdx.x >> 5);
  int tiles_n = N / (16*NB);
  int tiles = (M >> 4) * tiles_n;
  if (wave >= tiles) return;                    // wave-uniform exit, EXEC stays full
  int mt = wave / tiles_n, nt = wave % tiles_n;
  int lane = threadIdx.x & 31;
  int l15  = lane & 15;
  bool hi  = lane >= 16;
  const _Float16* Ap = A + (size_t)((mt << 4) + l15) * K + (hi ? 16 : 0);
  const _Float16* Wp[NB];
#pragma unroll
  for (int j = 0; j < NB; ++j)
    Wp[j] = W + (size_t)(((nt*NB + j) << 4) + l15) * K + (hi ? 16 : 0);
  __builtin_prefetch(Ap, 0, 0);
  v8f acc[NB] = {};
  for (int k0 = 0; k0 < K; k0 += 32) {
    v8h a0 = *(const v8h*)(Ap + k0);
    v8h a1 = *(const v8h*)(Ap + k0 + 8);
    v16h a;
#pragma unroll
    for (int i = 0; i < 8; ++i) { a[i] = a0[i]; a[8+i] = a1[i]; }
#pragma unroll
    for (int j = 0; j < NB; ++j) {
      v16h b = *(const v16h*)(Wp[j] + k0);
      acc[j] = __builtin_amdgcn_wmma_f32_16x16x32_f16(
          false, a, false, b, (short)0, acc[j], false, false);
    }
  }
#pragma unroll
  for (int j = 0; j < NB; ++j) {
    int ncol = ((nt*NB + j) << 4) + l15;
    float bv = bias ? bias[ncol] : 0.0f;
#pragma unroll
    for (int r = 0; r < 8; ++r) {
      int row = (mt << 4) + r + (hi ? 8 : 0);
      C[(size_t)row * N + ncol] = acc[j][r] + bv;
    }
  }
}

// ---------------------------------------------------------------------------
// Causal depthwise conv1d + SiLU, direction applied by index flip.
// ---------------------------------------------------------------------------
__global__ void conv_silu_kernel(const float* __restrict__ xz, const float* __restrict__ convw,
                                 const float* __restrict__ convb, _Float16* __restrict__ xih,
                                 int K, int dir) {
  int i = blockIdx.x*blockDim.x + threadIdx.x;   // over ROWS*D_INNER
  int d   = i & (D_INNER-1);
  int row = i >> 9;
  int l   = row & (SEQ-1);
  int b   = row >> 9;
  float acc = convb[d];
  for (int k = 0; k < K; ++k) {
    int ls = l - (K-1) + k;                      // direction-local source index
    if (ls >= 0) {
      int src = dir ? (SEQ-1-ls) : ls;
      acc += convw[d*K + k] * xz[((size_t)(b*SEQ + src))*(2*D_INNER) + d];
    }
  }
  float s = acc / (1.0f + __expf(-acc));         // SiLU
  xih[(size_t)i] = (_Float16)s;
}

// ---------------------------------------------------------------------------
// Chunked (3-phase) selective scan (8x parallelism over the serial scan).
// ---------------------------------------------------------------------------
__device__ __forceinline__ float softplusf(float x) {
  return (x > 20.0f) ? x : log1pf(__expf(x));
}
__device__ __forceinline__ float dt_of(const float* __restrict__ xd,
                                       const float* __restrict__ wdt, float dtbv) {
  float s = dtbv;
#pragma unroll
  for (int j = 0; j < DT_RANK; ++j) s += xd[j]*wdt[j];
  return softplusf(s);
}

__global__ void scan_pass1(const float* __restrict__ xdbl, const _Float16* __restrict__ xih,
                           const float* __restrict__ Alog, const float* __restrict__ Wdt,
                           const float* __restrict__ dtb,
                           float* __restrict__ Pa, float* __restrict__ Sh) {
  int tid = blockIdx.x*blockDim.x + threadIdx.x; // 0 .. BATCH*CHUNKS*D_INNER-1
  int d = tid & (D_INNER-1);
  int c = (tid >> 9) & (CHUNKS-1);
  int b = tid >> 12;
  float A[D_STATE], p[D_STATE], s[D_STATE], wdt[DT_RANK];
#pragma unroll
  for (int n = 0; n < D_STATE; ++n) { A[n] = -__expf(Alog[d*D_STATE + n]); p[n] = 1.0f; s[n] = 0.0f; }
#pragma unroll
  for (int j = 0; j < DT_RANK; ++j) wdt[j] = Wdt[d*DT_RANK + j];
  float dtbv = dtb[d];
  for (int i = 0; i < CLEN; ++i) {
    int row = b*SEQ + c*CLEN + i;
    const float* xd = xdbl + (size_t)row * XDBL_W;
    float dt  = dt_of(xd, wdt, dtbv);
    float dtx = dt * (float)xih[(size_t)row*D_INNER + d];
#pragma unroll
    for (int n = 0; n < D_STATE; ++n) {
      float a = __expf(dt * A[n]);
      s[n] = a*s[n] + dtx*xd[DT_RANK + n];
      p[n] *= a;
    }
  }
  size_t o = ((size_t)(b*D_INNER + d)*CHUNKS + c)*D_STATE;
#pragma unroll
  for (int n = 0; n < D_STATE; ++n) { Pa[o+n] = p[n]; Sh[o+n] = s[n]; }
}

__global__ void scan_combine(const float* __restrict__ Pa, const float* __restrict__ Sh,
                             float* __restrict__ Hst) {
  int tid = blockIdx.x*blockDim.x + threadIdx.x; // 0 .. BATCH*D_INNER-1
  float H[D_STATE];
#pragma unroll
  for (int n = 0; n < D_STATE; ++n) H[n] = 0.0f;
  for (int c = 0; c < CHUNKS; ++c) {
    size_t o = ((size_t)tid*CHUNKS + c)*D_STATE;
#pragma unroll
    for (int n = 0; n < D_STATE; ++n) {
      Hst[o+n] = H[n];                           // h at entry of chunk c
      H[n] = Pa[o+n]*H[n] + Sh[o+n];
    }
  }
}

__global__ void scan_pass2(const float* __restrict__ xz, const float* __restrict__ xdbl,
                           const _Float16* __restrict__ xih, const float* __restrict__ Alog,
                           const float* __restrict__ Dp, const float* __restrict__ Wdt,
                           const float* __restrict__ dtb, const float* __restrict__ Hst,
                           _Float16* __restrict__ ych, int dir) {
  int tid = blockIdx.x*blockDim.x + threadIdx.x;
  int d = tid & (D_INNER-1);
  int c = (tid >> 9) & (CHUNKS-1);
  int b = tid >> 12;
  float A[D_STATE], h[D_STATE], wdt[DT_RANK];
  size_t o = ((size_t)(b*D_INNER + d)*CHUNKS + c)*D_STATE;
#pragma unroll
  for (int n = 0; n < D_STATE; ++n) { A[n] = -__expf(Alog[d*D_STATE + n]); h[n] = Hst[o+n]; }
#pragma unroll
  for (int j = 0; j < DT_RANK; ++j) wdt[j] = Wdt[d*DT_RANK + j];
  float dpv = Dp[d], dtbv = dtb[d];
  for (int i = 0; i < CLEN; ++i) {
    int l = c*CLEN + i;
    int row = b*SEQ + l;
    const float* xd = xdbl + (size_t)row * XDBL_W;
    float dt  = dt_of(xd, wdt, dtbv);
    float xt  = (float)xih[(size_t)row*D_INNER + d];
    float dtx = dt * xt;
    float y = 0.0f;
#pragma unroll
    for (int n = 0; n < D_STATE; ++n) {
      float hn = __expf(dt * A[n]) * h[n] + dtx * xd[DT_RANK + n];
      h[n] = hn;
      y += hn * xd[DT_RANK + D_STATE + n];
    }
    int srcl = dir ? (SEQ-1-l) : l;
    float z = xz[((size_t)(b*SEQ + srcl))*(2*D_INNER) + D_INNER + d];
    float yc = (y + xt*dpv) * (z / (1.0f + __expf(-z)));
    ych[(size_t)row*D_INNER + d] = (_Float16)yc;
  }
}

// ---------------------------------------------------------------------------
__global__ void gateprep_kernel(const float* __restrict__ f00, const float* __restrict__ f10,
                                const float* __restrict__ f01, const float* __restrict__ f11,
                                float* __restrict__ ffwd, float* __restrict__ frev,
                                _Float16* __restrict__ cath) {
  int i = blockIdx.x*blockDim.x + threadIdx.x;   // over ROWS*D_MODEL
  int c   = i & (D_MODEL-1);
  int row = i >> 8;
  int l   = row & (SEQ-1);
  int b   = row >> 9;
  float fw = f00[i] + f10[i];
  size_t ri = (size_t)(b*SEQ + (SEQ-1-l))*D_MODEL + c;  // flip back
  float fr = f01[ri] + f11[ri];
  ffwd[i] = fw; frev[i] = fr;
  cath[(size_t)row*(2*D_MODEL) + c]           = (_Float16)fw;
  cath[(size_t)row*(2*D_MODEL) + D_MODEL + c] = (_Float16)fr;
}

__global__ void fuse_kernel(const float* __restrict__ glin, const float* __restrict__ gb,
                            const float* __restrict__ ffwd, const float* __restrict__ frev,
                            _Float16* __restrict__ fusedh) {
  int i = blockIdx.x*blockDim.x + threadIdx.x;   // over ROWS*D_MODEL
  int c = i & (D_MODEL-1);
  float g = 1.0f / (1.0f + __expf(-(glin[i] + gb[c])));
  fusedh[(size_t)i] = (_Float16)(g*ffwd[i] + (1.0f-g)*frev[i]);
}

// ---------------------------------------------------------------------------
extern "C" void kernel_launch(void* const* d_in, const int* in_sizes, int n_in,
                              void* d_out, int out_size, void* d_ws, size_t ws_size,
                              hipStream_t stream) {
  (void)in_sizes; (void)n_in; (void)out_size; (void)ws_size;
  const float* x    = (const float*)d_in[0];
  const float* ln_g = (const float*)d_in[1];
  const float* ln_b = (const float*)d_in[2];
  const float* Win[2]   = { (const float*)d_in[3],  (const float*)d_in[12] };
  const float* convw[2] = { (const float*)d_in[4],  (const float*)d_in[13] };
  const float* convb[2] = { (const float*)d_in[5],  (const float*)d_in[14] };
  const float* Wx[2]    = { (const float*)d_in[6],  (const float*)d_in[15] };
  const float* Wdt[2]   = { (const float*)d_in[7],  (const float*)d_in[16] };
  const float* dtb[2]   = { (const float*)d_in[8],  (const float*)d_in[17] };
  const float* Alog[2]  = { (const float*)d_in[9],  (const float*)d_in[18] };
  const float* Dp[2]    = { (const float*)d_in[10], (const float*)d_in[19] };
  const float* Wout[2]  = { (const float*)d_in[11], (const float*)d_in[20] };
  const int   convK[2]  = { 4, 8 };
  const float* gate_W = (const float*)d_in[21];
  const float* gate_b = (const float*)d_in[22];
  const float* op_W   = (const float*)d_in[23];
  const float* op_b   = (const float*)d_in[24];

  // ---- workspace carve-up (256B aligned sub-buffers) ----
  size_t off = 0;
  char* base = (char*)d_ws;
  auto carve = [&](size_t bytes) -> char* {
    char* p = base + off; off += (bytes + 255) & ~(size_t)255; return p;
  };
  float*    xn     = (float*)   carve((size_t)ROWS*D_MODEL*4);
  _Float16* xnh    = (_Float16*)carve((size_t)ROWS*D_MODEL*2);
  _Float16* winh[2]  = { (_Float16*)carve(2*D_INNER*D_MODEL*2), (_Float16*)carve(2*D_INNER*D_MODEL*2) };
  _Float16* wxh[2]   = { (_Float16*)carve(XDBL_W*D_INNER*2),    (_Float16*)carve(XDBL_W*D_INNER*2) };
  _Float16* wouth[2] = { (_Float16*)carve(D_MODEL*D_INNER*2),   (_Float16*)carve(D_MODEL*D_INNER*2) };
  _Float16* gwh    = (_Float16*)carve(D_MODEL*2*D_MODEL*2);
  _Float16* opwh   = (_Float16*)carve(D_MODEL*D_MODEL*2);
  float*    xz     = (float*)   carve((size_t)ROWS*2*D_INNER*4);   // reused per block
  _Float16* xih    = (_Float16*)carve((size_t)ROWS*D_INNER*2);     // reused per combo
  float*    xdbl   = (float*)   carve((size_t)ROWS*XDBL_W*4);      // reused per combo
  _Float16* ych    = (_Float16*)carve((size_t)ROWS*D_INNER*2);     // reused per combo
  float*    scanPa = (float*)   carve((size_t)BATCH*D_INNER*CHUNKS*D_STATE*4);
  float*    scanSh = (float*)   carve((size_t)BATCH*D_INNER*CHUNKS*D_STATE*4);
  float*    scanHs = (float*)   carve((size_t)BATCH*D_INNER*CHUNKS*D_STATE*4);
  float*    fpart[4];                                              // [dir*2+m]
  for (int i = 0; i < 4; ++i) fpart[i] = (float*)carve((size_t)ROWS*D_MODEL*4);
  float*    ffwd   = (float*)   carve((size_t)ROWS*D_MODEL*4);
  float*    frev   = (float*)   carve((size_t)ROWS*D_MODEL*4);
  _Float16* cath   = (_Float16*)carve((size_t)ROWS*2*D_MODEL*2);
  float*    glin   = (float*)   carve((size_t)ROWS*D_MODEL*4);
  _Float16* fusedh = (_Float16*)carve((size_t)ROWS*D_MODEL*2);

  // ---- 1) LayerNorm ----
  ln_kernel<<<ROWS, D_MODEL, 0, stream>>>(x, ln_g, ln_b, xn, xnh);

  // ---- 2) weight f32->f16 converts ----
  auto cvt = [&](const float* s, _Float16* d, int n) {
    f2h_kernel<<<(n + 255)/256, 256, 0, stream>>>(s, d, n);
  };
  for (int m = 0; m < 2; ++m) {
    cvt(Win[m],  winh[m],  2*D_INNER*D_MODEL);
    cvt(Wx[m],   wxh[m],   XDBL_W*D_INNER);
    cvt(Wout[m], wouth[m], D_MODEL*D_INNER);
  }
  cvt(gate_W, gwh,  D_MODEL*2*D_MODEL);
  cvt(op_W,   opwh, D_MODEL*D_MODEL);

  // Async-staged NB=4 kernel for N % 64 == 0; direct NB=5 for the N=80 GEMM.
  auto gemm = [&](const _Float16* A, const _Float16* W, float* C, const float* bias,
                  int M, int N, int K) {
    if (N == XDBL_W) {
      int tiles = M/16;
      wmma_gemm<5><<<(tiles + 3)/4, 128, 0, stream>>>(A, W, C, bias, M, N, K);
    } else {
      int blocks = (M/(16*GEMM_WAVES))*(N/64);
      wmma_gemm_async<4><<<blocks, 32*GEMM_WAVES, 0, stream>>>(A, W, C, bias, M, N, K);
    }
  };

  const int scan_threads = BATCH*CHUNKS*D_INNER;  // 32768

  // ---- 3) per sub-block, per direction ----
  for (int m = 0; m < 2; ++m) {
    gemm(xnh, winh[m], xz, nullptr, ROWS, 2*D_INNER, D_MODEL);  // shared by both dirs
    for (int dir = 0; dir < 2; ++dir) {
      conv_silu_kernel<<<(ROWS*D_INNER)/256, 256, 0, stream>>>(
          xz, convw[m], convb[m], xih, convK[m], dir);
      gemm(xih, wxh[m], xdbl, nullptr, ROWS, XDBL_W, D_INNER);
      scan_pass1<<<scan_threads/256, 256, 0, stream>>>(
          xdbl, xih, Alog[m], Wdt[m], dtb[m], scanPa, scanSh);
      scan_combine<<<(BATCH*D_INNER)/256, 256, 0, stream>>>(scanPa, scanSh, scanHs);
      scan_pass2<<<scan_threads/256, 256, 0, stream>>>(
          xz, xdbl, xih, Alog[m], Dp[m], Wdt[m], dtb[m], scanHs, ych, dir);
      gemm(ych, wouth[m], fpart[dir*2 + m], nullptr, ROWS, D_MODEL, D_INNER);
    }
  }

  // ---- 4) gate + fuse + output projection ----
  gateprep_kernel<<<(ROWS*D_MODEL)/256, 256, 0, stream>>>(
      fpart[0], fpart[1], fpart[2], fpart[3], ffwd, frev, cath);
  gemm(cath, gwh, glin, nullptr, ROWS, D_MODEL, 2*D_MODEL);
  fuse_kernel<<<(ROWS*D_MODEL)/256, 256, 0, stream>>>(glin, gate_b, ffwd, frev, fusedh);
  gemm(fusedh, opwh, (float*)d_out, op_b, ROWS, D_MODEL, D_MODEL);
}
